// ODENet_27230092657447
// MI455X (gfx1250) — matrix operand order repaired and checked by reading, not compiled, not run on hardware
//
#include <hip/hip_runtime.h>
#include <hip/hip_bf16.h>

typedef _Float16 half8  __attribute__((ext_vector_type(8)));
typedef _Float16 half16 __attribute__((ext_vector_type(16)));
typedef float    float8 __attribute__((ext_vector_type(8)));

#define T_STEPS 4096
#define BATCH   128
#define HDIM    256
#define TILE_B  16

#if defined(__has_builtin)
#if __has_builtin(__builtin_amdgcn_tanhf)
#define FAST_TANH(v) __builtin_amdgcn_tanhf(v)
#endif
#endif
#ifndef FAST_TANH
#define FAST_TANH(v) tanhf(v)
#endif

__global__ __launch_bounds__(256)
void odenet_euler_wmma(const float* __restrict__ x,
                       const float* __restrict__ W1,
                       const float* __restrict__ b1,
                       const float* __restrict__ W2,
                       const float* __restrict__ b2,
                       const float* __restrict__ W3,
                       const float* __restrict__ b3,
                       float* __restrict__ out)
{
    __shared__ __align__(16) _Float16 sH1[TILE_B * HDIM];  // 8 KB (A matrix, row major)
    __shared__ float sH2[TILE_B * HDIM];                   // 16 KB
    __shared__ float sW1y[HDIM], sW1x[HDIM], sB1[HDIM], sB2[HDIM], sW3[HDIM];
    __shared__ float sY[TILE_B];
    __shared__ float sPart[TILE_B * 16];

    const int tid = threadIdx.x;
    const int b0  = blockIdx.x * TILE_B;

    const int lane    = tid & 31;
    const int wave    = tid >> 5;
    const int n0      = wave * 32;                   // this wave's 32-column slab of H
    const int row     = tid >> 4;                    // 0..15 : batch row for layer1/3 work
    const int colBase = (tid & 15) << 4;             // 16 columns per thread

    // WMMA fragment addressing (wave32 CDNA5 layouts)
    const int aRow  = lane & 15;
    const int aOff  = (lane & 16) ? 8 : 0;           // A: lanes>=16 start at K+8
    const int bCol0 = n0 + (lane & 15);
    const int bOff  = (lane & 16) ? 16 : 0;          // B: lanes>=16 carry K 8..15,24..31
    const int mBase = (lane & 16) ? 8 : 0;           // D: lanes>=16 hold M 8..15

    // ---- stage small weights in LDS ----
    for (int i = tid; i < HDIM; i += 256) {
        sW1y[i] = W1[i];                             // weight on state (inp[:,0] = y)
        sW1x[i] = W1[HDIM + i];                      // weight on input sample
        sB1[i]  = b1[i];
        sB2[i]  = b2[i];
        sW3[i]  = W3[i];
    }
    if (tid < TILE_B) sY[tid] = 0.0f;
    const float bias3 = b3[0];

    // ---- build loop-invariant W2 B-fragments directly from global, keep in VGPRs ----
    // Fragment bf[2*kc + tile]: B 32x16 f16, lane carries column bCol0(+16),
    // halves j = K (kc*32 + bOff/2 + j) for j<8, K (kc*32 + 16 + bOff/2 + j-8) for j>=8.
    half16 bf[16];
    #pragma unroll
    for (int kc = 0; kc < 8; ++kc) {
        #pragma unroll
        for (int tile = 0; tile < 2; ++tile) {
            const int ncol = bCol0 + tile * 16;
            half16 v;
            #pragma unroll
            for (int j = 0; j < 16; ++j) {
                const int kk = kc * 32 + (j < 8 ? (bOff >> 1) + j : 16 + (bOff >> 1) + (j - 8));
                v[j] = (_Float16)W2[kk * HDIM + ncol];
            }
            bf[kc * 2 + tile] = v;
        }
    }
    __syncthreads();

    for (int t = 0; t < T_STEPS; ++t) {
        // ---- layer 1: h1 = tanh([y, x_t] @ W1 + b1), vectorized f16 stores ----
        const float yv = sY[row];
        const float xv = x[t * BATCH + b0 + row];
        if (t + 1 < T_STEPS && (tid & 15) == 0)
            __builtin_prefetch(&x[(t + 1) * BATCH + b0 + row], 0, 0);
        {
            union { half16 v; half8 h[2]; _Float16 e[16]; } hv;
            #pragma unroll
            for (int j = 0; j < 16; ++j) {
                const int c = colBase + j;
                hv.e[j] = (_Float16)FAST_TANH(fmaf(yv, sW1y[c], fmaf(xv, sW1x[c], sB1[c])));
            }
            _Float16* dst = &sH1[row * HDIM + colBase];
            *(half8*)dst       = hv.h[0];
            *(half8*)(dst + 8) = hv.h[1];
        }
        __syncthreads();

        // ---- layer 2: h2pre = h1 @ W2  via v_wmma_f32_16x16x32_f16 ----
        float8 acc0 = {}; float8 acc1 = {};
        #pragma unroll
        for (int kc = 0; kc < 8; ++kc) {
            const int k0 = kc * 32;
            union { half16 v; half8 h[2]; } a;
            const _Float16* ap = &sH1[aRow * HDIM + k0 + aOff];
            a.h[0] = *(const half8*)ap;              // K = base+0..7
            a.h[1] = *(const half8*)(ap + 16);       // K = base+16..23
            acc0 = __builtin_amdgcn_wmma_f32_16x16x32_f16(false, a.v, false, bf[kc * 2],
                                                          (short)0, acc0, false, false);
            acc1 = __builtin_amdgcn_wmma_f32_16x16x32_f16(false, a.v, false, bf[kc * 2 + 1],
                                                          (short)0, acc1, false, false);
        }

        // ---- bias + tanh, scatter D fragments to sH2 (f32) ----
        {
            const int nA = n0 + (lane & 15);
            const int nB = nA + 16;
            const float bA2 = sB2[nA], bB2 = sB2[nB];
            #pragma unroll
            for (int r = 0; r < 8; ++r) {
                sH2[(mBase + r) * HDIM + nA] = FAST_TANH(acc0[r] + bA2);
                sH2[(mBase + r) * HDIM + nB] = FAST_TANH(acc1[r] + bB2);
            }
        }
        __syncthreads();

        // ---- layer 3: dy = h2 @ W3 + b3  (two-level LDS reduction) ----
        {
            float p = 0.0f;
            #pragma unroll
            for (int j = 0; j < 16; ++j)
                p = fmaf(sH2[row * HDIM + colBase + j], sW3[colBase + j], p);
            sPart[row * 16 + (tid & 15)] = p;
        }
        __syncthreads();

        if (tid < TILE_B) {
            float dy = bias3;
            #pragma unroll
            for (int s = 0; s < 16; ++s) dy += sPart[tid * 16 + s];
            const float yo = sY[tid];
            out[t * BATCH + b0 + tid] = yo;          // scan emits pre-update state
            sY[tid] = yo + dy;                       // DT = 1.0
        }
        __syncthreads();
    }
}

extern "C" void kernel_launch(void* const* d_in, const int* in_sizes, int n_in,
                              void* d_out, int out_size, void* d_ws, size_t ws_size,
                              hipStream_t stream) {
    (void)in_sizes; (void)n_in; (void)out_size; (void)d_ws; (void)ws_size;
    const float* x  = (const float*)d_in[0];
    const float* W1 = (const float*)d_in[1];
    const float* b1 = (const float*)d_in[2];
    const float* W2 = (const float*)d_in[3];
    const float* b2 = (const float*)d_in[4];
    const float* W3 = (const float*)d_in[5];
    const float* b3 = (const float*)d_in[6];
    float* out = (float*)d_out;

    odenet_euler_wmma<<<dim3(BATCH / TILE_B), dim3(256), 0, stream>>>(
        x, W1, b1, W2, b2, W3, b3, out);
}
